// MultiHeadAttention_28063316312177
// MI455X (gfx1250) — matrix operand order repaired and checked by reading, not compiled, not run on hardware
//
#include <hip/hip_runtime.h>

typedef __attribute__((ext_vector_type(8)))  _Float16 v8h;
typedef __attribute__((ext_vector_type(16))) _Float16 v16h;
typedef __attribute__((ext_vector_type(8)))  float    v8f;

#define B_  2
#define L_  2048
#define D_  1024
#define H_  16
#define DK_ 64
#define M_  (B_ * L_)   // 4096 rows

#if __has_builtin(__builtin_amdgcn_tensor_load_to_lds) && \
    __has_builtin(__builtin_amdgcn_s_wait_tensorcnt)
#define USE_TDM 1
#else
#define USE_TDM 0
#endif

static __device__ __forceinline__ v16h combine8(v8h lo, v8h hi) {
  v16h r;
#pragma unroll
  for (int i = 0; i < 8; ++i) { r[i] = lo[i]; r[i + 8] = hi[i]; }
  return r;
}
static __device__ __forceinline__ v8f zero8() {
  v8f r;
#pragma unroll
  for (int i = 0; i < 8; ++i) r[i] = 0.0f;
  return r;
}
static __device__ __forceinline__ v8f wmma16(v16h a, v16h b, v8f c) {
  // D = A(16x32 f16) * B(32x16 f16) + C(16x16 f32)
  return __builtin_amdgcn_wmma_f32_16x16x32_f16(false, a, false, b, (short)0, c,
                                                false, false);
}

#if USE_TDM
typedef __attribute__((ext_vector_type(4))) unsigned int u32x4;
typedef __attribute__((ext_vector_type(8))) int i32x8;
typedef __attribute__((ext_vector_type(4))) int i32x4;
typedef __attribute__((address_space(3))) char* lds_raw_t;

static __device__ __forceinline__ unsigned lds_addr_of(void* p) {
  // generic (shared) pointer -> LDS byte offset (addrspace(3) is 32-bit)
  return (unsigned)(unsigned long long)(lds_raw_t)p;
}

// Tensor Data Mover: 2D tile of f16 (tile_d0 elems/row, tile_d1 rows,
// row stride stride0 elems) from global -> contiguous LDS rows.
static __device__ __forceinline__ void tdm_load_2d(unsigned lds_byte,
                                                   const void* gptr,
                                                   unsigned tile_d0,
                                                   unsigned tile_d1,
                                                   unsigned tensor_d0,
                                                   unsigned tensor_d1,
                                                   unsigned long long stride0) {
  const unsigned long long ga = (unsigned long long)gptr;
  u32x4 g0;
  g0[0] = 1u;                                               // count=1 (valid D#)
  g0[1] = lds_byte;                                         // lds_addr
  g0[2] = (unsigned)ga;                                     // global_addr[31:0]
  g0[3] = (unsigned)((ga >> 32) & 0x01FFFFFFu) | 0x80000000u; // [56:32] | type=2
  i32x8 g1;
  g1[0] = (int)(1u << 16);                                  // data_size=1 (2B)
  g1[1] = (int)((tensor_d0 & 0xFFFFu) << 16);               // tensor_dim0[15:0]
  g1[2] = (int)(((tensor_d0 >> 16) & 0xFFFFu) |             // tensor_dim0[31:16]
                ((tensor_d1 & 0xFFFFu) << 16));             // tensor_dim1[15:0]
  g1[3] = (int)(((tensor_d1 >> 16) & 0xFFFFu) |             // tensor_dim1[31:16]
                ((tile_d0 & 0xFFFFu) << 16));               // tile_dim0
  g1[4] = (int)(tile_d1 & 0xFFFFu);                         // tile_dim1 (dim2=0)
  g1[5] = (int)(stride0 & 0xFFFFFFFFull);                   // dim0_stride[31:0]
  g1[6] = (int)((stride0 >> 32) & 0xFFFFull);               // dim0_stride[47:32]
  g1[7] = 0;                                                // dim1_stride unused
  i32x4 z4 = {0, 0, 0, 0};                                  // 2D: groups 2/3 off
#if __has_include(<hip/amd_detail/amd_gfx1250_TDM.h>)
  i32x8 z8 = {0, 0, 0, 0, 0, 0, 0, 0};
  __builtin_amdgcn_tensor_load_to_lds(g0, g1, z4, z4, z8, 0);
#else
  __builtin_amdgcn_tensor_load_to_lds(g0, g1, z4, z4, 0);
#endif
}
#endif  // USE_TDM

// ---------------------------------------------------------------------------
// fp32 -> f16 conversion (grid-stride)
// ---------------------------------------------------------------------------
__global__ __launch_bounds__(256) void cvt_f32_f16(const float* __restrict__ in,
                                                   _Float16* __restrict__ out,
                                                   int n) {
  for (int i = blockIdx.x * blockDim.x + threadIdx.x; i < n;
       i += gridDim.x * blockDim.x)
    out[i] = (_Float16)in[i];
}

// ---------------------------------------------------------------------------
// y = A(f16, MxK row-major) @ W(f16, NxK row-major)^T + bias
//   mode 0: out = f32 [M][N]                     (final projection -> d_out)
//   mode 1: out = f16 at [(b*H+h)*L + l][dk], value scaled   (QKV layouts)
// Block tile 128x128, 8 waves of (32M x 64N), K-step 32.
// TDM double-buffered: wave 0 DMAs tile k+1 while all waves WMMA tile k.
// ---------------------------------------------------------------------------
__global__ __launch_bounds__(256) void gemm_wmma(const _Float16* __restrict__ A,
                                                 const _Float16* __restrict__ W,
                                                 const float* __restrict__ bias,
                                                 void* __restrict__ outp,
                                                 int mode, float scale) {
  __shared__ _Float16 sA[2][128][32];
  __shared__ _Float16 sB[2][128][32];

  const int tid  = threadIdx.x;
  const int wave = tid >> 5, lane = tid & 31;
  const int hi   = lane >> 4, lcol = lane & 15;
  const int wy   = wave & 3, wx = wave >> 2;           // 4x2 wave grid
  const int m0   = blockIdx.y * 128, n0 = blockIdx.x * 128;

  v8f acc[2][4];
#pragma unroll
  for (int mt = 0; mt < 2; ++mt)
#pragma unroll
    for (int nt = 0; nt < 4; ++nt) acc[mt][nt] = zero8();

  const _Float16* Abase = &A[(size_t)m0 * D_];
  const _Float16* Wbase = &W[(size_t)n0 * D_];

#if USE_TDM
  if (wave == 0) {
    tdm_load_2d(lds_addr_of(&sA[0][0][0]), Abase, 32, 128, D_, 4096, D_);
    tdm_load_2d(lds_addr_of(&sB[0][0][0]), Wbase, 32, 128, D_, 4096, D_);
    __builtin_amdgcn_s_wait_tensorcnt(0);
  }
  __syncthreads();
#endif

  for (int kb = 0; kb < D_ / 32; ++kb) {
#if USE_TDM
    const int cur = kb & 1;
    const bool more = (kb + 1 < D_ / 32);
    if (wave == 0 && more) {   // prefetch next tile via Tensor Data Mover
      tdm_load_2d(lds_addr_of(&sA[cur ^ 1][0][0]), Abase + (kb + 1) * 32,
                  32, 128, D_, 4096, D_);
      tdm_load_2d(lds_addr_of(&sB[cur ^ 1][0][0]), Wbase + (kb + 1) * 32,
                  32, 128, D_, 4096, D_);
    }
#else
    const int cur = 0;
    {   // synchronous staging fallback
      const int srow = tid >> 1;
      const int scol = (tid & 1) * 16;
      const int k0 = kb * 32;
      *(v8h*)&sA[0][srow][scol]     = *(const v8h*)&Abase[(size_t)srow * D_ + k0 + scol];
      *(v8h*)&sA[0][srow][scol + 8] = *(const v8h*)&Abase[(size_t)srow * D_ + k0 + scol + 8];
      *(v8h*)&sB[0][srow][scol]     = *(const v8h*)&Wbase[(size_t)srow * D_ + k0 + scol];
      *(v8h*)&sB[0][srow][scol + 8] = *(const v8h*)&Wbase[(size_t)srow * D_ + k0 + scol + 8];
      if (kb + 1 < D_ / 32) {
        __builtin_prefetch(&Abase[(size_t)srow * D_ + k0 + 32 + scol], 0, 1);
        __builtin_prefetch(&Wbase[(size_t)srow * D_ + k0 + 32 + scol], 0, 1);
      }
    }
    __syncthreads();
#endif

    // A fragments: lane m = lcol, K = e + 16*hi -> one contiguous 32B chunk
    v16h af[2];
#pragma unroll
    for (int mt = 0; mt < 2; ++mt) {
      const _Float16* ar = &sA[cur][wy * 32 + mt * 16 + lcol][0];
      af[mt] = *(const v16h*)&ar[16 * hi];
    }
    // B fragments: lane n = lcol, K = e + 16*hi -> one contiguous 32B chunk
#pragma unroll
    for (int nt = 0; nt < 4; ++nt) {
      const _Float16* br = &sB[cur][wx * 64 + nt * 16 + lcol][0];
      v16h bf = *(const v16h*)&br[16 * hi];
#pragma unroll
      for (int mt = 0; mt < 2; ++mt) acc[mt][nt] = wmma16(af[mt], bf, acc[mt][nt]);
    }

#if USE_TDM
    if (wave == 0 && more) __builtin_amdgcn_s_wait_tensorcnt(0);
#endif
    __syncthreads();
  }

  float bn[4];
#pragma unroll
  for (int nt = 0; nt < 4; ++nt) bn[nt] = bias[n0 + wx * 64 + nt * 16 + lcol];

  if (mode == 0) {
    float* out = (float*)outp;
#pragma unroll
    for (int mt = 0; mt < 2; ++mt)
#pragma unroll
      for (int nt = 0; nt < 4; ++nt)
#pragma unroll
        for (int r = 0; r < 8; ++r) {
          const int m = m0 + wy * 32 + mt * 16 + r + 8 * hi;
          const int n = n0 + wx * 64 + nt * 16 + lcol;
          out[(size_t)m * D_ + n] = acc[mt][nt][r] + bn[nt];
        }
  } else {
    _Float16* out = (_Float16*)outp;
#pragma unroll
    for (int mt = 0; mt < 2; ++mt)
#pragma unroll
      for (int nt = 0; nt < 4; ++nt)
#pragma unroll
        for (int r = 0; r < 8; ++r) {
          const int m  = m0 + wy * 32 + mt * 16 + r + 8 * hi;
          const int n  = n0 + wx * 64 + nt * 16 + lcol;
          const int b  = m >> 11, l = m & (L_ - 1);
          const int hh = n >> 6,  dk = n & (DK_ - 1);
          out[(((size_t)b * H_ + hh) * L_ + l) * DK_ + dk] =
              (_Float16)((acc[mt][nt][r] + bn[nt]) * scale);
        }
  }
}

// ---------------------------------------------------------------------------
// Flash-attention forward (causal). q is pre-scaled by 1/sqrt(DK).
// Grid: (L/128 query blocks, B*H). 8 waves, each owns 16 query rows.
// K tile (16 KB, contiguous) arrives via TDM while all threads stage V^T.
// ---------------------------------------------------------------------------
__global__ __launch_bounds__(256) void attn_fwd(const _Float16* __restrict__ qh,
                                                const _Float16* __restrict__ kh,
                                                const _Float16* __restrict__ vh,
                                                _Float16* __restrict__ atth) {
  __shared__ _Float16 sK[128][64];     // keys, row-major [key][dk]
  __shared__ _Float16 sVt[64][128];    // values transposed [dk][key]
  __shared__ _Float16 sP[8][16][32];   // per-wave P bounce (C->A layout)

  const int tid  = threadIdx.x;
  const int wave = tid >> 5, lane = tid & 31;
  const int hi   = lane >> 4, lcol = lane & 15;
  const int qblk = blockIdx.x;                 // 0..15
  const int bh   = blockIdx.y;                 // 0..31
  const int bidx = bh >> 4, h = bh & 15;
  const int qr0  = qblk * 128 + wave * 16;     // first query row of this wave

  // Q fragments: 16 rows x 64 dk -> two 16x32 A fragments, kept in VGPRs
  const _Float16* qrow = &qh[((size_t)bh * L_ + qr0 + lcol) * DK_];
  const v16h qa0 = *(const v16h*)&qrow[16 * hi];
  const v16h qa1 = *(const v16h*)&qrow[32 + 16 * hi];

  float mrow[8], lrow[8];
  v8f o[4];
#pragma unroll
  for (int r = 0; r < 8; ++r) { mrow[r] = -1e30f; lrow[r] = 0.0f; }
#pragma unroll
  for (int t = 0; t < 4; ++t) o[t] = zero8();

  const int rowmax = qr0 + 15;

  for (int kb = 0; kb <= qblk; ++kb) {
    const int kbase = kb * 128;
    // ---- stage K tile: 128x64 f16 is fully contiguous -> one 1D TDM ----
#if USE_TDM
    if (wave == 0)
      tdm_load_2d(lds_addr_of(&sK[0][0]), &kh[((size_t)bh * L_ + kbase) * DK_],
                  8192, 1, (unsigned)(L_ * DK_), 1, (unsigned long long)L_ * DK_);
#else
    {
      const int row = tid >> 1;
      const int ch  = (tid & 1) * 32;
      const _Float16* kg = &kh[((size_t)bh * L_ + kbase + row) * DK_ + ch];
#pragma unroll
      for (int j = 0; j < 4; ++j) *(v8h*)&sK[row][ch + 8 * j] = *(const v8h*)&kg[8 * j];
    }
#endif
    // ---- stage V^T tile (manual: needs transpose) ----
#pragma unroll
    for (int c = 0; c < 4; ++c) {
      const int chunk = tid + 256 * c;      // 0..1023
      const int key = chunk >> 3;
      const int db  = (chunk & 7) * 8;
      v8h vv = *(const v8h*)&vh[((size_t)bh * L_ + kbase + key) * DK_ + db];
#pragma unroll
      for (int j = 0; j < 8; ++j) sVt[db + j][key] = vv[j];
    }
#if USE_TDM
    if (wave == 0) __builtin_amdgcn_s_wait_tensorcnt(0);
#endif
    __syncthreads();

    for (int sub = 0; sub < 4; ++sub) {
      const int k0 = kbase + sub * 32;
      if (k0 > rowmax) break;               // causal skip (wave-uniform)

      // ---- S(16x32) = Q(16x64) @ K^T : 2 tiles x 2 K-steps ----
      v8f s[2];
#pragma unroll
      for (int t = 0; t < 2; ++t) {
        const int krow = sub * 32 + t * 16 + lcol;       // B: N = key row
        v16h b0 = *(const v16h*)&sK[krow][16 * hi];      // dk 0..31
        v16h b1 = *(const v16h*)&sK[krow][32 + 16 * hi]; // dk 32..63
        v8f c = wmma16(qa0, b0, zero8());
        s[t] = wmma16(qa1, b1, c);
      }
      if (kb == qblk) {                     // causal mask on diagonal block
#pragma unroll
        for (int t = 0; t < 2; ++t) {
          const int n = k0 + t * 16 + lcol;
#pragma unroll
          for (int r = 0; r < 8; ++r)
            if (n > qr0 + r + 8 * hi) s[t][r] = -1e30f;
        }
      }

      // ---- online softmax (rows live across 16 lanes of each half) ----
      float p0[8], p1[8];
#pragma unroll
      for (int r = 0; r < 8; ++r) {
        float tmax = fmaxf(s[0][r], s[1][r]);
#pragma unroll
        for (int off = 1; off < 16; off <<= 1)
          tmax = fmaxf(tmax, __shfl_xor(tmax, off, 32));
        const float mn    = fmaxf(mrow[r], tmax);
        const float alpha = __expf(mrow[r] - mn);
        mrow[r] = mn;
        p0[r] = __expf(s[0][r] - mn);
        p1[r] = __expf(s[1][r] - mn);
        float ps = p0[r] + p1[r];
#pragma unroll
        for (int off = 1; off < 16; off <<= 1) ps += __shfl_xor(ps, off, 32);
        lrow[r] = lrow[r] * alpha + ps;
#pragma unroll
        for (int t = 0; t < 4; ++t) o[t][r] *= alpha;
      }

      // ---- P: C layout -> A layout via per-wave LDS bounce ----
      _Float16* pw = &sP[wave][0][0];
#pragma unroll
      for (int r = 0; r < 8; ++r) {
        const int m = r + 8 * hi;
        pw[m * 32 + lcol]      = (_Float16)p0[r];
        pw[m * 32 + 16 + lcol] = (_Float16)p1[r];
      }
      // DS ops are in-order within a wave -> RAW through LDS is safe
      v16h pa = *(const v16h*)&pw[lcol * 32 + 16 * hi];

      // ---- O(16x64) += P(16x32) @ V(32x64) ----
#pragma unroll
      for (int t = 0; t < 4; ++t) {
        v16h vb = *(const v16h*)&sVt[t * 16 + lcol][sub * 32 + 16 * hi];
        o[t] = wmma16(pa, vb, o[t]);
      }
    }
    __syncthreads();
  }

  // normalize and store f16 at [b][l][h*64+dk] (row-major input of O-proj)
#pragma unroll
  for (int r = 0; r < 8; ++r) {
    const float inv = 1.0f / lrow[r];
    const int m = qr0 + r + 8 * hi;
    _Float16* orow = &atth[((size_t)bidx * L_ + m) * D_ + h * DK_];
#pragma unroll
    for (int t = 0; t < 4; ++t) orow[t * 16 + lcol] = (_Float16)(o[t][r] * inv);
  }
}

// ---------------------------------------------------------------------------
extern "C" void kernel_launch(void* const* d_in, const int* in_sizes, int n_in,
                              void* d_out, int out_size, void* d_ws, size_t ws_size,
                              hipStream_t stream) {
  (void)in_sizes; (void)n_in; (void)out_size; (void)ws_size;
  const float* x  = (const float*)d_in[0];
  const float* wq = (const float*)d_in[1];
  const float* bq = (const float*)d_in[2];
  const float* wk = (const float*)d_in[3];
  const float* bk = (const float*)d_in[4];
  const float* wv = (const float*)d_in[5];
  const float* bv = (const float*)d_in[6];
  const float* wo = (const float*)d_in[7];
  const float* bo = (const float*)d_in[8];

  char* ws = (char*)d_ws;
  size_t off = 0;
  _Float16* xh   = (_Float16*)(ws + off); off += (size_t)M_ * D_ * 2;   // 8 MiB
  _Float16* wqh  = (_Float16*)(ws + off); off += (size_t)D_ * D_ * 2;   // 2 MiB
  _Float16* wkh  = (_Float16*)(ws + off); off += (size_t)D_ * D_ * 2;
  _Float16* wvh  = (_Float16*)(ws + off); off += (size_t)D_ * D_ * 2;
  _Float16* woh  = (_Float16*)(ws + off); off += (size_t)D_ * D_ * 2;
  _Float16* qhp  = (_Float16*)(ws + off); off += (size_t)M_ * D_ * 2;   // [b][h][l][dk]
  _Float16* khp  = (_Float16*)(ws + off); off += (size_t)M_ * D_ * 2;
  _Float16* vhp  = (_Float16*)(ws + off); off += (size_t)M_ * D_ * 2;
  _Float16* atth = (_Float16*)(ws + off); off += (size_t)M_ * D_ * 2;
  // total ~50 MiB of workspace

  const int nx = M_ * D_;   // 4,194,304
  const int nw = D_ * D_;   // 1,048,576
  cvt_f32_f16<<<2048, 256, 0, stream>>>(x,  xh,  nx);
  cvt_f32_f16<<<1024, 256, 0, stream>>>(wq, wqh, nw);
  cvt_f32_f16<<<1024, 256, 0, stream>>>(wk, wkh, nw);
  cvt_f32_f16<<<1024, 256, 0, stream>>>(wv, wvh, nw);
  cvt_f32_f16<<<1024, 256, 0, stream>>>(wo, woh, nw);

  dim3 gg(D_ / 128, M_ / 128);   // (8, 32)
  gemm_wmma<<<gg, 256, 0, stream>>>(xh, wqh, bq, qhp, 1, 0.125f);  // Q * 1/sqrt(64)
  gemm_wmma<<<gg, 256, 0, stream>>>(xh, wkh, bk, khp, 1, 1.0f);
  gemm_wmma<<<gg, 256, 0, stream>>>(xh, wvh, bv, vhp, 1, 1.0f);

  attn_fwd<<<dim3(L_ / 128, B_ * H_), 256, 0, stream>>>(qhp, khp, vhp, atth);

  gemm_wmma<<<gg, 256, 0, stream>>>(atth, woh, bo, d_out, 0, 1.0f);
}
